// en_gnn_88347477279281
// MI455X (gfx1250) — matrix operand order repaired and checked by reading, not compiled, not run on hardware
//
#include <hip/hip_runtime.h>
#include <hip/hip_bf16.h>
#include <math.h>
#include <stdint.h>

#define NN 30000
#define EE 480000
#define GG 128
#define NCLS 55
#define LLAYERS 4

typedef __attribute__((ext_vector_type(16))) _Float16 v16h;
typedef __attribute__((ext_vector_type(8)))  _Float16 v8h;
typedef __attribute__((ext_vector_type(4)))  _Float16 v4h;
typedef __attribute__((ext_vector_type(8)))  float    v8f;

// ---- WMMA fragment helpers (CDNA5 ISA 7.12.2 layouts, wave32) ----
__device__ __forceinline__ v16h frag_a(const _Float16* p) {
  v8h lo = *(const v8h*)(p);
  v8h hi = *(const v8h*)(p + 16);
  return __builtin_shufflevector(lo, hi, 0, 1, 2, 3, 4, 5, 6, 7,
                                 8, 9, 10, 11, 12, 13, 14, 15);
}

__device__ __forceinline__ v16h frag_bf(const _Float16* sWf, int kt, int nt, int lane) {
  return *(const v16h*)(sWf + (((kt * 4 + nt) * 32 + lane) << 4));
}

// Stage a [KT*32, 64] f32 weight matrix into fragment-ready f16 LDS layout.
template <int KT>
__device__ __forceinline__ void stage_weights(const float* __restrict__ W,
                                              _Float16* sWf, int tid) {
  for (int fi = tid; fi < KT * 4 * 32; fi += 128) {
    int lane2 = fi & 31;
    int nt = (fi >> 5) & 3;
    int kt = fi >> 7;
    int n = nt * 16 + (lane2 & 15);
    int kb = kt * 32 + ((lane2 & 16) ? 8 : 0);
    v16h w;
#pragma unroll
    for (int e = 0; e < 16; ++e) w[e] = (_Float16)W[(kb + (e < 8 ? e : e + 8)) * 64 + n];
    *(v16h*)(sWf + ((size_t)fi << 4)) = w;
  }
}

// 4x f32 -> 4x f16 via cvt_pkrtz (round toward zero), single 8B store (works for LDS or global dst)
__device__ __forceinline__ void store4h(_Float16* dst, float4 f) {
  auto lo = __builtin_amdgcn_cvt_pkrtz(f.x, f.y);
  auto hi = __builtin_amdgcn_cvt_pkrtz(f.z, f.w);
  uint2 u;
  u.x = __builtin_bit_cast(unsigned, lo);
  u.y = __builtin_bit_cast(unsigned, hi);
  *(uint2*)dst = u;
}

// CDNA5 async global->LDS DMA (16B per lane), tracked by ASYNCcnt.
__device__ __forceinline__ void async_copy16(void* lds_dst, const void* gsrc) {
  unsigned lds = (unsigned)(uintptr_t)lds_dst;            // low 32 bits = LDS address
  unsigned long long ga = (unsigned long long)(uintptr_t)gsrc;
  asm volatile("global_load_async_to_lds_b128 %0, %1, off"
               :: "v"(lds), "v"(ga) : "memory");
}
__device__ __forceinline__ void async_wait0() {
  asm volatile("s_wait_asynccnt 0x0" ::: "memory");
}

#define WMMA_F16(A, B, C) \
  __builtin_amdgcn_wmma_f32_16x16x32_f16(false, (A), false, (B), (short)0, (C), false, false)

__device__ __forceinline__ float silu(float v) { return v / (1.0f + __expf(-v)); }

__device__ __forceinline__ unsigned ford(float f) {
  unsigned u = __float_as_uint(f);
  return (u & 0x80000000u) ? ~u : (u | 0x80000000u);
}
__device__ __forceinline__ float funord(unsigned u) {
  return __uint_as_float((u & 0x80000000u) ? (u & 0x7FFFFFFFu) : ~u);
}

// ---------------- embedding_in: h0[N,32] @ W[32,64] + b -> h (f16) ; copy x ----------------
__global__ __launch_bounds__(128) void egnn_emb_in_kernel(
    const float* __restrict__ h0, const float* __restrict__ x0,
    const float* __restrict__ W, const float* __restrict__ b,
    _Float16* __restrict__ h, float* __restrict__ x) {
  __shared__ __align__(32) _Float16 sWf[1 * 4 * 32 * 16];
  __shared__ __align__(32) _Float16 sA[64 * 32];
  __shared__ __align__(16) float sD[64 * 64];
  __shared__ float sB[64];
  const int tid = threadIdx.x, lane = tid & 31, wv = tid >> 5;
  stage_weights<1>(W, sWf, tid);
  if (tid < 64) sB[tid] = b[tid];
  const int n0 = blockIdx.x * 64;
  {
    int ln = tid >> 1, half = tid & 1;
    int nn = n0 + ln;
    _Float16* dst = sA + ln * 32 + half * 16;
    if (nn < NN) {
      const float4* s4 = (const float4*)(h0 + (size_t)nn * 32 + half * 16);
#pragma unroll
      for (int k = 0; k < 4; ++k) store4h(dst + k * 4, s4[k]);
      if (!half) {
        x[nn * 3 + 0] = x0[nn * 3 + 0];
        x[nn * 3 + 1] = x0[nn * 3 + 1];
        x[nn * 3 + 2] = x0[nn * 3 + 2];
      }
    } else {
      v4h z = {};
#pragma unroll
      for (int k = 0; k < 4; ++k) *(v4h*)(dst + k * 4) = z;
    }
  }
  __syncthreads();
  const int mrow = lane & 15, hi8 = (lane & 16) ? 8 : 0;
  v8f acc[4] = {};
  v16h a = frag_a(sA + (wv * 16 + mrow) * 32 + hi8);
#pragma unroll
  for (int nt = 0; nt < 4; ++nt) acc[nt] = WMMA_F16(a, frag_bf(sWf, 0, nt, lane), acc[nt]);
#pragma unroll
  for (int nt = 0; nt < 4; ++nt) {
    int n = nt * 16 + (lane & 15);
    float bb = sB[n];
#pragma unroll
    for (int r = 0; r < 8; ++r) sD[(wv * 16 + r + hi8) * 64 + n] = acc[nt][r] + bb;
  }
  // vectorized f16 store: thread reads rows written by its own wave -> no barrier
  {
    int rr = tid >> 1, cb = (tid & 1) * 32;
    int nn = n0 + rr;
    if (nn < NN) {
      const float4* sp = (const float4*)(sD + rr * 64 + cb);
      _Float16* dp = h + (size_t)nn * 64 + cb;
#pragma unroll
      for (int k = 0; k < 8; ++k) store4h(dp + k * 4, sp[k]);
    }
  }
}

// ---------------- fused per-layer edge kernel ----------------
__global__ __launch_bounds__(128) void egnn_edge_kernel(
    const _Float16* __restrict__ h, const float* __restrict__ x,
    const int* __restrict__ erow, const int* __restrict__ ecol,
    const float* __restrict__ W1, const float* __restrict__ b1,
    const float* __restrict__ W2, const float* __restrict__ b2,
    const float* __restrict__ Wc1, const float* __restrict__ bc1,
    const float* __restrict__ wc2,
    float* __restrict__ tsum, float* __restrict__ tcnt, float* __restrict__ agg) {
  __shared__ __align__(32) _Float16 sW1f[4 * 4 * 32 * 16];
  __shared__ __align__(32) _Float16 sW2f[2 * 4 * 32 * 16];
  __shared__ __align__(32) _Float16 sWc1f[2 * 4 * 32 * 16];
  __shared__ __align__(32) _Float16 sA[64 * 128];   // 64-edge tile; reused per stage
  __shared__ float sW1r[64], sB1[64], sB2[64], sBc1[64], sWc2[64];
  __shared__ float sRad[64], sDiff[64 * 3];
  __shared__ int sRow[64];

  const int tid = threadIdx.x, lane = tid & 31, wv = tid >> 5;

  stage_weights<4>(W1, sW1f, tid);    // K rows 0..127 (h_i, h_j)
  stage_weights<2>(W2, sW2f, tid);
  stage_weights<2>(Wc1, sWc1f, tid);
  if (tid < 64) {
    sW1r[tid] = W1[128 * 64 + tid];   // radial row (k = 2H)
    sB1[tid] = b1[tid]; sB2[tid] = b2[tid]; sBc1[tid] = bc1[tid]; sWc2[tid] = wc2[tid];
  }

  const int e0 = blockIdx.x * 64;     // E is a multiple of 64: no tail guards
  {
    int le = tid >> 1, half = tid & 1;
    int e = e0 + le;
    int r = erow[e], c = ecol[e];
    const _Float16* hs = h + (size_t)(half ? c : r) * 64;   // f16 row: pure byte copy
    _Float16* dst = sA + le * 128 + half * 64;
#pragma unroll
    for (int k = 0; k < 8; ++k) async_copy16(dst + k * 8, hs + k * 8);
    if (!half) {
      sRow[le] = r;
      float dx = x[r * 3 + 0] - x[c * 3 + 0];
      float dy = x[r * 3 + 1] - x[c * 3 + 1];
      float dz = x[r * 3 + 2] - x[c * 3 + 2];
      sDiff[le * 3 + 0] = dx; sDiff[le * 3 + 1] = dy; sDiff[le * 3 + 2] = dz;
      sRad[le] = dx * dx + dy * dy + dz * dz;
    }
  }
  async_wait0();
  __syncthreads();

  const int mrow = lane & 15, hi8 = (lane & 16) ? 8 : 0;
  const _Float16* Abase = sA + (wv * 16 + mrow) * 128;

  // stage 1: silu(eat @ W1 + radial*w1r + b1) -> cols 0..63
  v8f acc[4] = {};
#pragma unroll
  for (int kt = 0; kt < 4; ++kt) {
    v16h a = frag_a(Abase + kt * 32 + hi8);
#pragma unroll
    for (int nt = 0; nt < 4; ++nt) acc[nt] = WMMA_F16(a, frag_bf(sW1f, kt, nt, lane), acc[nt]);
  }
  float radv[8];
#pragma unroll
  for (int r = 0; r < 8; ++r) radv[r] = sRad[wv * 16 + r + hi8];
#pragma unroll
  for (int nt = 0; nt < 4; ++nt) {
    int n = nt * 16 + (lane & 15);
    float wr = sW1r[n], bb = sB1[n];
#pragma unroll
    for (int r = 0; r < 8; ++r) {
      int le = wv * 16 + r + hi8;
      sA[le * 128 + n] = (_Float16)silu(acc[nt][r] + radv[r] * wr + bb);
    }
  }

  // stage 2: m = silu(t @ W2 + b2) -> cols 64..127
  v8f acc2[4] = {};
#pragma unroll
  for (int kt = 0; kt < 2; ++kt) {
    v16h a = frag_a(Abase + kt * 32 + hi8);
#pragma unroll
    for (int nt = 0; nt < 4; ++nt) acc2[nt] = WMMA_F16(a, frag_bf(sW2f, kt, nt, lane), acc2[nt]);
  }
#pragma unroll
  for (int nt = 0; nt < 4; ++nt) {
    int n = nt * 16 + (lane & 15);
    float bb = sB2[n];
#pragma unroll
    for (int r = 0; r < 8; ++r) {
      int le = wv * 16 + r + hi8;
      sA[le * 128 + 64 + n] = (_Float16)silu(acc2[nt][r] + bb);
    }
  }

  // stage 3: p = silu(m @ Wc1 + bc1) -> cols 0..63
  v8f acc3[4] = {};
#pragma unroll
  for (int kt = 0; kt < 2; ++kt) {
    v16h a = frag_a(Abase + 64 + kt * 32 + hi8);
#pragma unroll
    for (int nt = 0; nt < 4; ++nt) acc3[nt] = WMMA_F16(a, frag_bf(sWc1f, kt, nt, lane), acc3[nt]);
  }
#pragma unroll
  for (int nt = 0; nt < 4; ++nt) {
    int n = nt * 16 + (lane & 15);
    float bb = sBc1[n];
#pragma unroll
    for (int r = 0; r < 8; ++r) {
      int le = wv * 16 + r + hi8;
      sA[le * 128 + n] = (_Float16)silu(acc3[nt][r] + bb);
    }
  }

  // c = p . wc2 (linear), coordinate scatter
  if (lane < 16) {
    int le = wv * 16 + lane;
    const _Float16* pr = sA + le * 128;
    float s = 0.0f;
#pragma unroll
    for (int k0 = 0; k0 < 64; k0 += 8) {
      v8h c8 = *(const v8h*)(pr + k0);
      float4 w0 = *(const float4*)(sWc2 + k0);
      float4 w1 = *(const float4*)(sWc2 + k0 + 4);
      s += (float)c8[0] * w0.x + (float)c8[1] * w0.y + (float)c8[2] * w0.z + (float)c8[3] * w0.w;
      s += (float)c8[4] * w1.x + (float)c8[5] * w1.y + (float)c8[6] * w1.z + (float)c8[7] * w1.w;
    }
    int rn = sRow[le];
    atomicAdd(&tsum[rn * 3 + 0], sDiff[le * 3 + 0] * s);
    atomicAdd(&tsum[rn * 3 + 1], sDiff[le * 3 + 1] * s);
    atomicAdd(&tsum[rn * 3 + 2], sDiff[le * 3 + 2] * s);
    atomicAdd(&tcnt[rn], 1.0f);
  }
  // message aggregation: agg[row] += m (cols 64..127); lane -> (row, 32-col slab)
  {
    int mr = lane >> 1;
    int cb = (lane & 1) * 32;
    int le = wv * 16 + mr;
    int rn = sRow[le];
    const _Float16* pm = sA + le * 128 + 64 + cb;
    float* dp = agg + (size_t)rn * 64 + cb;
#pragma unroll
    for (int k0 = 0; k0 < 32; k0 += 8) {
      v8h m8 = *(const v8h*)(pm + k0);
#pragma unroll
      for (int i = 0; i < 8; ++i) atomicAdd(dp + k0 + i, (float)m8[i]);
    }
  }
}

// ---------------- per-layer node kernel: coord update + node MLP ----------------
__global__ __launch_bounds__(128) void egnn_node_kernel(
    _Float16* __restrict__ h, float* __restrict__ x,
    const float* __restrict__ tsum, const float* __restrict__ tcnt,
    const float* __restrict__ agg,
    const float* __restrict__ W1, const float* __restrict__ b1,
    const float* __restrict__ W2, const float* __restrict__ b2) {
  __shared__ __align__(32) _Float16 sW1f[4 * 4 * 32 * 16];
  __shared__ __align__(32) _Float16 sW2f[2 * 4 * 32 * 16];
  __shared__ __align__(32) _Float16 sA[64 * 128];
  __shared__ __align__(16) float sD[64 * 64];
  __shared__ float sB1[64], sB2[64];
  const int tid = threadIdx.x, lane = tid & 31, wv = tid >> 5;
  stage_weights<4>(W1, sW1f, tid);
  stage_weights<2>(W2, sW2f, tid);
  if (tid < 64) { sB1[tid] = b1[tid]; sB2[tid] = b2[tid]; }

  const int n0 = blockIdx.x * 64;
  {
    int ln = tid >> 1, half = tid & 1;
    int nn = n0 + ln;
    _Float16* dst = sA + ln * 128 + half * 64;
    if (nn < NN) {
      if (half) {  // agg (f32) -> f16
        const float4* s4 = (const float4*)(agg + (size_t)nn * 64);
#pragma unroll
        for (int k = 0; k < 16; ++k) store4h(dst + k * 4, s4[k]);
      } else {     // h already f16: async byte copy
        const _Float16* hs = h + (size_t)nn * 64;
#pragma unroll
        for (int k = 0; k < 8; ++k) async_copy16(dst + k * 8, hs + k * 8);
        float c = tcnt[nn]; c = c < 1.0f ? 1.0f : c;  // x += t_sum / max(cnt,1)
        x[nn * 3 + 0] += tsum[nn * 3 + 0] / c;
        x[nn * 3 + 1] += tsum[nn * 3 + 1] / c;
        x[nn * 3 + 2] += tsum[nn * 3 + 2] / c;
      }
    } else {
      v4h z = {};
#pragma unroll
      for (int k = 0; k < 16; ++k) *(v4h*)(dst + k * 4) = z;
    }
  }
  async_wait0();
  __syncthreads();

  const int mrow = lane & 15, hi8 = (lane & 16) ? 8 : 0;
  const _Float16* Abase = sA + (wv * 16 + mrow) * 128;

  v8f acc[4] = {};
#pragma unroll
  for (int kt = 0; kt < 4; ++kt) {
    v16h a = frag_a(Abase + kt * 32 + hi8);
#pragma unroll
    for (int nt = 0; nt < 4; ++nt) acc[nt] = WMMA_F16(a, frag_bf(sW1f, kt, nt, lane), acc[nt]);
  }
#pragma unroll
  for (int nt = 0; nt < 4; ++nt) {
    int n = nt * 16 + (lane & 15);
    float bb = sB1[n];
#pragma unroll
    for (int r = 0; r < 8; ++r) {
      int ln = wv * 16 + r + hi8;
      sA[ln * 128 + n] = (_Float16)silu(acc[nt][r] + bb);
    }
  }
  v8f acc2[4] = {};
#pragma unroll
  for (int kt = 0; kt < 2; ++kt) {
    v16h a = frag_a(Abase + kt * 32 + hi8);
#pragma unroll
    for (int nt = 0; nt < 4; ++nt) acc2[nt] = WMMA_F16(a, frag_bf(sW2f, kt, nt, lane), acc2[nt]);
  }
#pragma unroll
  for (int nt = 0; nt < 4; ++nt) {
    int n = nt * 16 + (lane & 15);
    float bb = sB2[n];
#pragma unroll
    for (int r = 0; r < 8; ++r) sD[(wv * 16 + r + hi8) * 64 + n] = acc2[nt][r] + bb;
  }
  // vectorized f16 store of new h (rows are wave-local -> no barrier)
  {
    int rr = tid >> 1, cb = (tid & 1) * 32;
    int nn = n0 + rr;
    if (nn < NN) {
      const float4* sp = (const float4*)(sD + rr * 64 + cb);
      _Float16* dp = h + (size_t)nn * 64 + cb;
#pragma unroll
      for (int k = 0; k < 8; ++k) store4h(dp + k * 4, sp[k]);
    }
  }
}

// ---------------- embedding_out + segment-max pooling ----------------
__global__ __launch_bounds__(128) void egnn_emb_out_kernel(
    const _Float16* __restrict__ h, const int* __restrict__ batch,
    const float* __restrict__ W, const float* __restrict__ b,
    unsigned* __restrict__ hpu) {
  __shared__ __align__(32) _Float16 sWf[2 * 4 * 32 * 16];
  __shared__ __align__(32) _Float16 sA[64 * 64];
  __shared__ float sB[64];
  const int tid = threadIdx.x, lane = tid & 31, wv = tid >> 5;
  stage_weights<2>(W, sWf, tid);
  if (tid < 64) sB[tid] = b[tid];
  const int n0 = blockIdx.x * 64;
  {
    int ln = tid >> 1, half = tid & 1;
    int nn = n0 + ln;
    _Float16* dst = sA + ln * 64 + half * 32;
    if (nn < NN) {
      const _Float16* hs = h + (size_t)nn * 64 + half * 32;
#pragma unroll
      for (int k = 0; k < 4; ++k) async_copy16(dst + k * 8, hs + k * 8);
    } else {
      v4h z = {};
#pragma unroll
      for (int k = 0; k < 8; ++k) *(v4h*)(dst + k * 4) = z;
    }
  }
  async_wait0();
  __syncthreads();
  const int mrow = lane & 15, hi8 = (lane & 16) ? 8 : 0;
  const _Float16* Abase = sA + (wv * 16 + mrow) * 64;
  v8f acc[4] = {};
#pragma unroll
  for (int kt = 0; kt < 2; ++kt) {
    v16h a = frag_a(Abase + kt * 32 + hi8);
#pragma unroll
    for (int nt = 0; nt < 4; ++nt) acc[nt] = WMMA_F16(a, frag_bf(sWf, kt, nt, lane), acc[nt]);
  }
  float bb[4];
#pragma unroll
  for (int nt = 0; nt < 4; ++nt) bb[nt] = sB[nt * 16 + (lane & 15)];
#pragma unroll
  for (int r = 0; r < 8; ++r) {  // row-outer: one guard + one batch load per row
    int nn = n0 + wv * 16 + r + hi8;
    if (nn < NN) {
      int gb = batch[nn] * 64;
#pragma unroll
      for (int nt = 0; nt < 4; ++nt) {
        int n = nt * 16 + (lane & 15);
        atomicMax(&hpu[gb + n], ford(acc[nt][r] + bb[nt]));
      }
    }
  }
}

// ---------------- small helper kernels ----------------
__global__ void egnn_zero_acc(float* __restrict__ tsum, float* __restrict__ tcnt,
                              float4* __restrict__ agg4) {
  int i = blockIdx.x * blockDim.x + threadIdx.x;
  float4 z4 = {0.0f, 0.0f, 0.0f, 0.0f};
  if (i < NN * 16) agg4[i] = z4;   // N*64 floats as float4
  if (i < NN * 3) tsum[i] = 0.0f;
  if (i < NN) tcnt[i] = 0.0f;
}

__global__ void egnn_init_hp(unsigned* __restrict__ hpu) {
  int i = blockIdx.x * blockDim.x + threadIdx.x;
  if (i < GG * 64) hpu[i] = ford(-3.402823466e38f);
}

__global__ void egnn_hp_unmap(const unsigned* __restrict__ hpu, float* __restrict__ hpf) {
  int i = blockIdx.x * blockDim.x + threadIdx.x;
  if (i < GG * 64) hpf[i] = funord(hpu[i]);
}

__global__ void egnn_fc(const float* __restrict__ in, const float* __restrict__ W,
                        const float* __restrict__ b, float* __restrict__ out,
                        int In, int On, int relu) {
  int i = blockIdx.x * blockDim.x + threadIdx.x;
  if (i >= GG * On) return;
  int g = i / On, o = i - g * On;
  float s = b[o];
  for (int k = 0; k < In; ++k) s += in[g * In + k] * W[k * On + o];
  out[i] = relu ? fmaxf(s, 0.0f) : s;
}

__global__ void egnn_logsoftmax(const float* __restrict__ logits, float* __restrict__ out) {
  int g = blockIdx.x;
  int lane = threadIdx.x;  // one wave32
  float mx = -3.402823466e38f;
  for (int k = lane; k < NCLS; k += 32) mx = fmaxf(mx, logits[g * NCLS + k]);
  for (int off = 16; off; off >>= 1) mx = fmaxf(mx, __shfl_xor(mx, off));
  float sum = 0.0f;
  for (int k = lane; k < NCLS; k += 32) sum += __expf(logits[g * NCLS + k] - mx);
  for (int off = 16; off; off >>= 1) sum += __shfl_xor(sum, off);
  float lse = mx + __logf(sum);
  for (int k = lane; k < NCLS; k += 32) out[g * NCLS + k] = logits[g * NCLS + k] - lse;
}

// ---------------- host launch ----------------
extern "C" void kernel_launch(void* const* d_in, const int* in_sizes, int n_in,
                              void* d_out, int out_size, void* d_ws, size_t ws_size,
                              hipStream_t stream) {
  (void)in_sizes; (void)n_in; (void)out_size; (void)ws_size;
  const float* h0       = (const float*)d_in[0];
  const float* x0       = (const float*)d_in[1];
  const int*   ei       = (const int*)d_in[2];
  const int*   bat      = (const int*)d_in[3];
  const float* emb_in_w = (const float*)d_in[4];
  const float* emb_in_b = (const float*)d_in[5];
  const float* edge_w1  = (const float*)d_in[6];
  const float* edge_b1  = (const float*)d_in[7];
  const float* edge_w2  = (const float*)d_in[8];
  const float* edge_b2  = (const float*)d_in[9];
  const float* node_w1  = (const float*)d_in[10];
  const float* node_b1  = (const float*)d_in[11];
  const float* node_w2  = (const float*)d_in[12];
  const float* node_b2  = (const float*)d_in[13];
  const float* coord_w1 = (const float*)d_in[14];
  const float* coord_b1 = (const float*)d_in[15];
  const float* coord_w2 = (const float*)d_in[16];
  const float* emb_out_w= (const float*)d_in[17];
  const float* emb_out_b= (const float*)d_in[18];
  const float* fc1_w    = (const float*)d_in[19];
  const float* fc1_b    = (const float*)d_in[20];
  const float* fc2_w    = (const float*)d_in[21];
  const float* fc2_b    = (const float*)d_in[22];
  const float* fc3_w    = (const float*)d_in[23];
  const float* fc3_b    = (const float*)d_in[24];

  _Float16* hbuf = (_Float16*)d_ws;               // [N,64] f16 (3.84 MB)
  float* xbuf = (float*)(hbuf + (size_t)NN * 64); // [N,3]
  float* tsum = xbuf + (size_t)NN * 3;            // [N,3]
  float* tcnt = tsum + (size_t)NN * 3;            // [N]
  float* aggb = tcnt + NN;                        // [N,64] f32
  unsigned* hpu = (unsigned*)(aggb + (size_t)NN * 64);  // [G,64] ordered-uint
  float* hpf = (float*)(hpu + GG * 64);           // [G,64]
  float* z1  = hpf + GG * 64;                     // [G,128]
  float* z2  = z1 + GG * 128;                     // [G,128]
  float* lg  = z2 + GG * 128;                     // [G,55]

  const int* erow = ei;
  const int* ecol = ei + EE;

  dim3 blk(128);
  const int nodeBlocks = (NN + 63) / 64;
  const int edgeBlocks = EE / 64;  // exact

  egnn_emb_in_kernel<<<nodeBlocks, blk, 0, stream>>>(h0, x0, emb_in_w, emb_in_b, hbuf, xbuf);

  for (int l = 0; l < LLAYERS; ++l) {
    egnn_zero_acc<<<(NN * 64 + 255) / 256, 256, 0, stream>>>(tsum, tcnt, (float4*)aggb);
    egnn_edge_kernel<<<edgeBlocks, blk, 0, stream>>>(
        hbuf, xbuf, erow, ecol,
        edge_w1 + (size_t)l * 129 * 64, edge_b1 + l * 64,
        edge_w2 + (size_t)l * 64 * 64,  edge_b2 + l * 64,
        coord_w1 + (size_t)l * 64 * 64, coord_b1 + l * 64,
        coord_w2 + (size_t)l * 64,
        tsum, tcnt, aggb);
    egnn_node_kernel<<<nodeBlocks, blk, 0, stream>>>(
        hbuf, xbuf, tsum, tcnt, aggb,
        node_w1 + (size_t)l * 128 * 64, node_b1 + l * 64,
        node_w2 + (size_t)l * 64 * 64,  node_b2 + l * 64);
  }

  egnn_init_hp<<<(GG * 64 + 255) / 256, 256, 0, stream>>>(hpu);
  egnn_emb_out_kernel<<<nodeBlocks, blk, 0, stream>>>(hbuf, bat, emb_out_w, emb_out_b, hpu);
  egnn_hp_unmap<<<(GG * 64 + 255) / 256, 256, 0, stream>>>(hpu, hpf);
  egnn_fc<<<(GG * 128 + 255) / 256, 256, 0, stream>>>(hpf, fc1_w, fc1_b, z1, 64, 128, 1);
  egnn_fc<<<(GG * 128 + 255) / 256, 256, 0, stream>>>(z1, fc2_w, fc2_b, z2, 128, 128, 1);
  egnn_fc<<<(GG * NCLS + 255) / 256, 256, 0, stream>>>(z2, fc3_w, fc3_b, lg, 128, NCLS, 0);
  egnn_logsoftmax<<<GG, 32, 0, stream>>>(lg, (float*)d_out);
}
